// AttentionDecoder_17051020165700
// MI455X (gfx1250) — compile-verified, matrix operand order checked by reading, not run on hardware
//
#include <hip/hip_runtime.h>
#include <hip/hip_bf16.h>
#include <math.h>

// ---------------- problem constants (match reference) ----------------
#define BB    64
#define TT    150
#define LCTX  256
#define HH    512
#define EE    512
#define VV    1000
#define VPAD  1024
#define G4    2048   // 4*H
#define KCAT  1024   // E + H

typedef __attribute__((ext_vector_type(16))) __bf16 v16bf;
typedef __attribute__((ext_vector_type(8)))  __bf16 v8bf;
typedef __attribute__((ext_vector_type(8)))  float  v8f;

// ---------------- workspace layout (bytes), all 256B-aligned ----------------
static const size_t OFF_WCAT   = 0;                                   // 2048*1024 bf16
static const size_t OFF_WOM    = OFF_WCAT   + (size_t)G4*KCAT*2;      // 512*1024 bf16
static const size_t OFF_WLOGIT = OFF_WOM    + (size_t)HH*KCAT*2;      // 1024*512 bf16 (padded)
static const size_t OFF_BIASG  = OFF_WLOGIT + (size_t)VPAD*HH*2;      // 2048 f32
static const size_t OFF_BLOGIT = OFF_BIASG  + (size_t)G4*4;           // 1024 f32
static const size_t OFF_XALL   = OFF_BLOGIT + (size_t)VPAD*4;         // B*T*E bf16
static const size_t OFF_H      = OFF_XALL   + (size_t)BB*TT*EE*2;     // B*H f32
static const size_t OFF_C      = OFF_H      + (size_t)BB*HH*4;
static const size_t OFF_HBF    = OFF_C      + (size_t)BB*HH*4;        // B*H bf16
static const size_t OFF_CATBF  = OFF_HBF    + (size_t)BB*HH*2;        // B*1024 bf16 [ctx|h2]
static const size_t OFF_GATES  = OFF_CATBF  + (size_t)BB*KCAT*2;      // B*2048 f32
static const size_t OFF_OUTBF  = OFF_GATES  + (size_t)BB*G4*4;        // B*H bf16
static const size_t OFF_LOGITS = OFF_OUTBF  + (size_t)BB*HH*2;        // B*1024 f32
static const size_t OFF_WHMT   = OFF_LOGITS + (size_t)BB*VPAD*4;      // 512*512 f32 (W_hm^T)

// ---------------- device helpers ----------------
__device__ __forceinline__ unsigned short f2bf(float f) {
    unsigned int u = __float_as_uint(f);
    u += 0x7FFFu + ((u >> 16) & 1u);          // round-to-nearest-even
    return (unsigned short)(u >> 16);
}
__device__ __forceinline__ float sigm(float x) { return 1.0f / (1.0f + __expf(-x)); }

// Per-lane 16-elem bf16 fragment = two contiguous 8-elem runs (two b128 loads).
// Matches 16-bit A/B VGPR layout: lanes 0-15 hold K=0-7 & 16-23, lanes 16-31 hold K=8-15 & 24-31.
__device__ __forceinline__ v16bf ld16(const __bf16* p) {
    v8bf lo = *reinterpret_cast<const v8bf*>(p);
    v8bf hi = *reinterpret_cast<const v8bf*>(p + 16);
    v16bf r;
#pragma unroll
    for (int i = 0; i < 8; ++i) { r[i] = lo[i]; r[i + 8] = hi[i]; }
    return r;
}

// ---------------- prep kernels (once per launch) ----------------
__global__ void k_prep_wcat(const float* W_ih, const float* W_hh, unsigned short* wcat) {
    size_t idx = (size_t)blockIdx.x * blockDim.x + threadIdx.x;
    if (idx >= (size_t)G4 * KCAT) return;
    int g = (int)(idx / KCAT), k = (int)(idx % KCAT);
    float v = (k < EE) ? W_ih[(size_t)g * EE + k] : W_hh[(size_t)g * HH + (k - EE)];
    wcat[idx] = f2bf(v);
}
__global__ void k_prep_wom(const float* W_om, unsigned short* wom) {
    size_t idx = (size_t)blockIdx.x * blockDim.x + threadIdx.x;
    if (idx >= (size_t)HH * KCAT) return;
    wom[idx] = f2bf(W_om[idx]);
}
__global__ void k_prep_wlogit(const float* W_logit, unsigned short* wlg) {
    size_t idx = (size_t)blockIdx.x * blockDim.x + threadIdx.x;
    if (idx >= (size_t)VPAD * HH) return;
    int row = (int)(idx / HH), col = (int)(idx % HH);
    float v = (row < VV) ? W_logit[(size_t)row * HH + col] : 0.0f;
    wlg[idx] = f2bf(v);
}
__global__ void k_prep_whmt(const float* W_hm, float* whmt) {
    size_t idx = (size_t)blockIdx.x * blockDim.x + threadIdx.x;
    if (idx >= (size_t)HH * HH) return;
    int k = (int)(idx / HH), i = (int)(idx % HH);
    whmt[idx] = W_hm[(size_t)i * HH + k];     // whmt[k][i] = W_hm[i][k]
}
__global__ void k_prep_misc(const float* b_ih, const float* b_hh, const float* b_logit,
                            float* biasg, float* blgp, float* h, float* c) {
    int idx = blockIdx.x * blockDim.x + threadIdx.x;
    if (idx < G4)   biasg[idx] = b_ih[idx] + b_hh[idx];
    if (idx < VPAD) blgp[idx]  = (idx < VV) ? b_logit[idx] : 0.0f;
    if (idx < BB * HH) { h[idx] = 0.0f; c[idx] = 0.0f; }
}
__global__ void k_prep_x(const int* seq, const float* embed, unsigned short* xall) {
    size_t idx = (size_t)blockIdx.x * blockDim.x + threadIdx.x;
    if (idx >= (size_t)BB * TT * EE) return;
    int b = (int)(idx / ((size_t)TT * EE));
    int rem = (int)(idx - (size_t)b * TT * EE);
    int t = rem / EE, k = rem % EE;
    int tok = seq[b * TT + t];
    float v = embed[(size_t)tok * EE + k];
    xall[idx] = f2bf(v > 0.0f ? v : 0.0f);   // relu(embed)
}

// ---------------- WMMA GEMM, 2x2 register blocking (32x32 per wave) ----------------
// gates = [x_t | h] @ Wcat^T + bias ; grid: (8, 2), block 256 (8 waves)
// K loop split into two compile-time halves (x-source, h-source): no conditionals inside.
__global__ __launch_bounds__(256) void k_gates(const unsigned short* xall,
                                               const unsigned short* hbf,
                                               const unsigned short* wcat,
                                               const float* biasg,
                                               float* gates, int t) {
    const int lane = threadIdx.x & 31, wv = threadIdx.x >> 5;
    const int n0 = (blockIdx.x * 8 + wv) * 32;
    const int m0 = blockIdx.y * 32;
    const int lm = lane & 15;
    const int hob = (lane >> 4) * 8;   // fold into base pointers
    const __bf16* pX0 = reinterpret_cast<const __bf16*>(xall) + ((size_t)(m0 + lm) * TT + t) * EE + hob;
    const __bf16* pX1 = reinterpret_cast<const __bf16*>(xall) + ((size_t)(m0 + 16 + lm) * TT + t) * EE + hob;
    const __bf16* pH0 = reinterpret_cast<const __bf16*>(hbf) + (size_t)(m0 + lm) * HH + hob;
    const __bf16* pH1 = reinterpret_cast<const __bf16*>(hbf) + (size_t)(m0 + 16 + lm) * HH + hob;
    const __bf16* pB0 = reinterpret_cast<const __bf16*>(wcat) + (size_t)(n0 + lm) * KCAT + hob;
    const __bf16* pB1 = reinterpret_cast<const __bf16*>(wcat) + (size_t)(n0 + 16 + lm) * KCAT + hob;

    v8f a00, a01, a10, a11;
#pragma unroll
    for (int i = 0; i < 8; ++i) { a00[i] = 0.f; a01[i] = 0.f; a10[i] = 0.f; a11[i] = 0.f; }

    // phase 1: k in [0, EE) from x_t
    for (int kt = 0; kt < EE; kt += 32) {
        v16bf fa0 = ld16(pX0 + kt);
        v16bf fa1 = ld16(pX1 + kt);
        v16bf fb0 = ld16(pB0 + kt);
        v16bf fb1 = ld16(pB1 + kt);
        a00 = __builtin_amdgcn_wmma_f32_16x16x32_bf16(false, fa0, false, fb0, (short)0, a00, false, false);
        a01 = __builtin_amdgcn_wmma_f32_16x16x32_bf16(false, fa0, false, fb1, (short)0, a01, false, false);
        a10 = __builtin_amdgcn_wmma_f32_16x16x32_bf16(false, fa1, false, fb0, (short)0, a10, false, false);
        a11 = __builtin_amdgcn_wmma_f32_16x16x32_bf16(false, fa1, false, fb1, (short)0, a11, false, false);
    }
    // phase 2: k in [EE, KCAT) from h
    for (int kt = 0; kt < HH; kt += 32) {
        v16bf fa0 = ld16(pH0 + kt);
        v16bf fa1 = ld16(pH1 + kt);
        v16bf fb0 = ld16(pB0 + EE + kt);
        v16bf fb1 = ld16(pB1 + EE + kt);
        a00 = __builtin_amdgcn_wmma_f32_16x16x32_bf16(false, fa0, false, fb0, (short)0, a00, false, false);
        a01 = __builtin_amdgcn_wmma_f32_16x16x32_bf16(false, fa0, false, fb1, (short)0, a01, false, false);
        a10 = __builtin_amdgcn_wmma_f32_16x16x32_bf16(false, fa1, false, fb0, (short)0, a10, false, false);
        a11 = __builtin_amdgcn_wmma_f32_16x16x32_bf16(false, fa1, false, fb1, (short)0, a11, false, false);
    }

    const int c0 = n0 + lm, c1 = n0 + 16 + lm;
    const float bs0 = biasg[c0], bs1 = biasg[c1];
    const int hor = (lane >> 4) * 8;
#pragma unroll
    for (int r = 0; r < 8; ++r) {
        int r0 = m0 + hor + r, r1 = m0 + 16 + hor + r;
        gates[(size_t)r0 * G4 + c0] = a00[r] + bs0;
        gates[(size_t)r0 * G4 + c1] = a01[r] + bs1;
        gates[(size_t)r1 * G4 + c0] = a10[r] + bs0;
        gates[(size_t)r1 * G4 + c1] = a11[r] + bs1;
    }
}

// generic: C = A[MxK] @ B[NxK]^T (+bias), K compile-time ; grid: (N/256, M/32), block 256
template <int K>
__global__ __launch_bounds__(256) void k_gemm(const unsigned short* A, const unsigned short* Bw,
                                              const float* bias, float* Cf, unsigned short* Cbf,
                                              int N) {
    const int lane = threadIdx.x & 31, wv = threadIdx.x >> 5;
    const int n0 = (blockIdx.x * 8 + wv) * 32;
    const int m0 = blockIdx.y * 32;
    const int lm = lane & 15;
    const int hob = (lane >> 4) * 8;
    const __bf16* pA0 = reinterpret_cast<const __bf16*>(A) + (size_t)(m0 + lm) * K + hob;
    const __bf16* pA1 = reinterpret_cast<const __bf16*>(A) + (size_t)(m0 + 16 + lm) * K + hob;
    const __bf16* pB0 = reinterpret_cast<const __bf16*>(Bw) + (size_t)(n0 + lm) * K + hob;
    const __bf16* pB1 = reinterpret_cast<const __bf16*>(Bw) + (size_t)(n0 + 16 + lm) * K + hob;

    v8f a00, a01, a10, a11;
#pragma unroll
    for (int i = 0; i < 8; ++i) { a00[i] = 0.f; a01[i] = 0.f; a10[i] = 0.f; a11[i] = 0.f; }

    for (int kt = 0; kt < K; kt += 32) {
        v16bf fa0 = ld16(pA0 + kt);
        v16bf fa1 = ld16(pA1 + kt);
        v16bf fb0 = ld16(pB0 + kt);
        v16bf fb1 = ld16(pB1 + kt);
        a00 = __builtin_amdgcn_wmma_f32_16x16x32_bf16(false, fa0, false, fb0, (short)0, a00, false, false);
        a01 = __builtin_amdgcn_wmma_f32_16x16x32_bf16(false, fa0, false, fb1, (short)0, a01, false, false);
        a10 = __builtin_amdgcn_wmma_f32_16x16x32_bf16(false, fa1, false, fb0, (short)0, a10, false, false);
        a11 = __builtin_amdgcn_wmma_f32_16x16x32_bf16(false, fa1, false, fb1, (short)0, a11, false, false);
    }

    const int c0 = n0 + lm, c1 = n0 + 16 + lm;
    const float bs0 = bias ? bias[c0] : 0.0f;
    const float bs1 = bias ? bias[c1] : 0.0f;
    const int hor = (lane >> 4) * 8;
#pragma unroll
    for (int r = 0; r < 8; ++r) {
        int r0 = m0 + hor + r, r1 = m0 + 16 + hor + r;
        float v00 = a00[r] + bs0, v01 = a01[r] + bs1;
        float v10 = a10[r] + bs0, v11 = a11[r] + bs1;
        if (Cf) {
            Cf[(size_t)r0 * N + c0] = v00; Cf[(size_t)r0 * N + c1] = v01;
            Cf[(size_t)r1 * N + c0] = v10; Cf[(size_t)r1 * N + c1] = v11;
        }
        if (Cbf) {
            Cbf[(size_t)r0 * N + c0] = f2bf(v00); Cbf[(size_t)r0 * N + c1] = f2bf(v01);
            Cbf[(size_t)r1 * N + c0] = f2bf(v10); Cbf[(size_t)r1 * N + c1] = f2bf(v11);
        }
    }
}

// ---------------- LSTM cell + attention, one block per batch element ----------------
__global__ __launch_bounds__(256) void k_cell_attn(const float* gates, float* h, float* c,
                                                   unsigned short* hbf, unsigned short* catbf,
                                                   const float* whmt, const float* cnn) {
    const int b = blockIdx.x, tid = threadIdx.x;
    __shared__ float sh_h[HH];
    __shared__ float sh_m[HH];
    __shared__ float sh_w[LCTX];
    __shared__ float sred[256];

    // phase 1: LSTM cell
    for (int j = tid; j < HH; j += 256) {
        float gi = gates[(size_t)b * G4 + j];
        float gf = gates[(size_t)b * G4 + HH + j];
        float gg = gates[(size_t)b * G4 + 2 * HH + j];
        float go = gates[(size_t)b * G4 + 3 * HH + j];
        float cp = c[(size_t)b * HH + j];
        float c2 = sigm(gf) * cp + sigm(gi) * tanhf(gg);
        float h2 = sigm(go) * tanhf(c2);
        c[(size_t)b * HH + j] = c2;
        h[(size_t)b * HH + j] = h2;
        sh_h[j] = h2;
        unsigned short hb = f2bf(h2);
        hbf[(size_t)b * HH + j] = hb;
        catbf[(size_t)b * KCAT + HH + j] = hb;   // second half of [ctx|h2]
    }
    __syncthreads();

    // phase 2: mapped = h2 @ W_hm^T  via transposed weights (coalesced)
    {
        float m0a = 0.0f, m1a = 0.0f;
        for (int k = 0; k < HH; ++k) {
            float hv = sh_h[k];
            const float* row = whmt + (size_t)k * HH;
            m0a += hv * row[tid];
            m1a += hv * row[tid + 256];
        }
        sh_m[tid] = m0a;
        sh_m[tid + 256] = m1a;
    }
    __syncthreads();

    // phase 3: attention scores + softmax (tid == l, LCTX == 256)
    const float* cf = cnn + ((size_t)b * LCTX + tid) * HH;
    float s = 0.0f;
    for (int k = 0; k < HH; ++k) s += cf[k] * sh_m[k];
    sred[tid] = s;
    __syncthreads();
    for (int off = 128; off > 0; off >>= 1) {
        if (tid < off) sred[tid] = fmaxf(sred[tid], sred[tid + off]);
        __syncthreads();
    }
    const float mx = sred[0];
    __syncthreads();
    float e = __expf(s - mx);
    sred[tid] = e;
    __syncthreads();
    for (int off = 128; off > 0; off >>= 1) {
        if (tid < off) sred[tid] += sred[tid + off];
        __syncthreads();
    }
    const float inv = 1.0f / sred[0];
    __syncthreads();
    sh_w[tid] = e * inv;
    __syncthreads();

    // phase 4: ctx = w^T @ cnn_feats[b]
    for (int hh = tid; hh < HH; hh += 256) {
        float a = 0.0f;
        for (int l = 0; l < LCTX; ++l)
            a += sh_w[l] * cnn[((size_t)b * LCTX + l) * HH + hh];
        catbf[(size_t)b * KCAT + hh] = f2bf(a);  // first half of [ctx|h2]
    }
}

// ---------------- log-softmax over V=1000 and scatter to output ----------------
__global__ __launch_bounds__(256) void k_logsoftmax(const float* logits, float* out, int t) {
    const int b = blockIdx.x, tid = threadIdx.x;
    __shared__ float sred[256];
    const float* lg = logits + (size_t)b * VPAD;

    float mx = -3.4e38f;
    for (int v = tid; v < VV; v += 256) mx = fmaxf(mx, lg[v]);
    sred[tid] = mx;
    __syncthreads();
    for (int off = 128; off > 0; off >>= 1) {
        if (tid < off) sred[tid] = fmaxf(sred[tid], sred[tid + off]);
        __syncthreads();
    }
    mx = sred[0];
    __syncthreads();
    float s = 0.0f;
    for (int v = tid; v < VV; v += 256) s += __expf(lg[v] - mx);
    sred[tid] = s;
    __syncthreads();
    for (int off = 128; off > 0; off >>= 1) {
        if (tid < off) sred[tid] += sred[tid + off];
        __syncthreads();
    }
    const float lse = mx + __logf(sred[0]);
    float* dst = out + ((size_t)b * TT + t) * VV;
    for (int v = tid; v < VV; v += 256) dst[v] = lg[v] - lse;
}

// ---------------- host launcher ----------------
extern "C" void kernel_launch(void* const* d_in, const int* in_sizes, int n_in,
                              void* d_out, int out_size, void* d_ws, size_t ws_size,
                              hipStream_t stream) {
    const float* cnn      = (const float*)d_in[0];
    const int*   seq      = (const int*)d_in[1];
    const float* embed    = (const float*)d_in[2];
    const float* W_ih     = (const float*)d_in[3];
    const float* b_ih     = (const float*)d_in[4];
    const float* W_hh     = (const float*)d_in[5];
    const float* b_hh     = (const float*)d_in[6];
    const float* W_hm     = (const float*)d_in[7];
    const float* W_om     = (const float*)d_in[8];
    const float* W_logit  = (const float*)d_in[9];
    const float* b_logit  = (const float*)d_in[10];
    float* out = (float*)d_out;

    char* ws = (char*)d_ws;
    unsigned short* wcat   = (unsigned short*)(ws + OFF_WCAT);
    unsigned short* wom    = (unsigned short*)(ws + OFF_WOM);
    unsigned short* wlg    = (unsigned short*)(ws + OFF_WLOGIT);
    float*          biasg  = (float*)(ws + OFF_BIASG);
    float*          blgp   = (float*)(ws + OFF_BLOGIT);
    unsigned short* xall   = (unsigned short*)(ws + OFF_XALL);
    float*          h      = (float*)(ws + OFF_H);
    float*          c      = (float*)(ws + OFF_C);
    unsigned short* hbf    = (unsigned short*)(ws + OFF_HBF);
    unsigned short* catbf  = (unsigned short*)(ws + OFF_CATBF);
    float*          gates  = (float*)(ws + OFF_GATES);
    unsigned short* outbf  = (unsigned short*)(ws + OFF_OUTBF);
    float*          logits = (float*)(ws + OFF_LOGITS);
    float*          whmt   = (float*)(ws + OFF_WHMT);

    // one-time prep (per launch; deterministic)
    {
        size_t n;
        n = (size_t)G4 * KCAT;
        k_prep_wcat<<<dim3((unsigned)((n + 255) / 256)), 256, 0, stream>>>(W_ih, W_hh, wcat);
        n = (size_t)HH * KCAT;
        k_prep_wom<<<dim3((unsigned)((n + 255) / 256)), 256, 0, stream>>>(W_om, wom);
        n = (size_t)VPAD * HH;
        k_prep_wlogit<<<dim3((unsigned)((n + 255) / 256)), 256, 0, stream>>>(W_logit, wlg);
        n = (size_t)HH * HH;
        k_prep_whmt<<<dim3((unsigned)((n + 255) / 256)), 256, 0, stream>>>(W_hm, whmt);
        k_prep_misc<<<dim3((BB * HH + 255) / 256), 256, 0, stream>>>(b_ih, b_hh, b_logit,
                                                                     biasg, blgp, h, c);
        n = (size_t)BB * TT * EE;
        k_prep_x<<<dim3((unsigned)((n + 255) / 256)), 256, 0, stream>>>(seq, embed, xall);
    }

    // sequential decode: 5 kernels per step
    for (int t = 0; t < TT; ++t) {
        k_gates<<<dim3(G4 / 256, BB / 32), 256, 0, stream>>>(xall, hbf, wcat, biasg, gates, t);
        k_cell_attn<<<dim3(BB), 256, 0, stream>>>(gates, h, c, hbf, catbf, whmt, cnn);
        // out = [ctx|h2] @ W_om^T  -> bf16
        k_gemm<KCAT><<<dim3(HH / 256, BB / 32), 256, 0, stream>>>(catbf, wom, nullptr,
                                                                  nullptr, outbf, HH);
        // logits = out @ W_logit^T + b_logit  -> f32 (padded N=1024)
        k_gemm<HH><<<dim3(VPAD / 256, BB / 32), 256, 0, stream>>>(outbf, wlg, blgp,
                                                                  logits, nullptr, VPAD);
        k_logsoftmax<<<dim3(BB), 256, 0, stream>>>(logits, out, t);
    }
}
